// SparseEdgeUpdateLayer_4784593568415
// MI455X (gfx1250) — matrix operand order repaired and checked by reading, not compiled, not run on hardware
//
#include <hip/hip_runtime.h>
#include <hip/hip_bf16.h>

typedef _Float16 v16h __attribute__((ext_vector_type(16)));
typedef _Float16 v8h  __attribute__((ext_vector_type(8)));
typedef _Float16 v4h  __attribute__((ext_vector_type(4)));
typedef float    v8f  __attribute__((ext_vector_type(8)));

#define IN_DIM   192
#define EDGE_DIM 64
#define NODE_DIM 64
#define XPITCH   200   // f16 pitch for x / h-f16 rows (400 B, 16B aligned)
#define HPITCH   196   // f32 pitch for h rows
#define ETILE    32    // edges per block-iteration

// d_ws layout: [0, 73728) W1 packed f16, [73728, 98304) W2 packed f16.
// Packed layout per (ktile, ntile) block of 512 f16:
//   lane l (0..31) holds 16 contiguous f16 = W[k0 + (j>>3)*16 + (l>>4)*8 + (j&7)][n0 + (l&15)], j=0..15
// which is exactly the per-lane B fragment of V_WMMA_F32_16X16X32_F16.
__global__ __launch_bounds__(256) void pack_weights(const float* __restrict__ W1,
                                                    const float* __restrict__ W2,
                                                    _Float16* __restrict__ w1p,
                                                    _Float16* __restrict__ w2p) {
    int idx = blockIdx.x * 256 + threadIdx.x;
    if (idx < IN_DIM * IN_DIM) {          // 36864
        int b = idx >> 9, r = idx & 511, l = r >> 4, j = r & 15;
        int kt = b / 12, nt = b % 12;
        int k = kt * 32 + ((j >> 3) << 4) + ((l >> 4) << 3) + (j & 7);
        int n = nt * 16 + (l & 15);
        w1p[idx] = (_Float16)W1[k * IN_DIM + n];
    }
    if (idx < IN_DIM * EDGE_DIM) {        // 12288
        int b = idx >> 9, r = idx & 511, l = r >> 4, j = r & 15;
        int kt = b >> 2, nt = b & 3;
        int k = kt * 32 + ((j >> 3) << 4) + ((l >> 4) << 3) + (j & 7);
        int n = nt * 16 + (l & 15);
        w2p[idx] = (_Float16)W2[k * EDGE_DIM + n];
    }
}

__global__ __launch_bounds__(256) void edge_update_kernel(
        const float* __restrict__ nodef, const float* __restrict__ edgef,
        const int* __restrict__ eidx,
        const float* __restrict__ b1, const float* __restrict__ gamma,
        const float* __restrict__ beta, const float* __restrict__ b2,
        const _Float16* __restrict__ w1p, const _Float16* __restrict__ w2p,
        float* __restrict__ out, int nEdges, int nTiles) {
    extern __shared__ char smem[];
    _Float16* sW1 = (_Float16*)(smem);                 // 73728 B
    _Float16* sW2 = (_Float16*)(smem + 73728);         // 24576 B
    _Float16* sX  = (_Float16*)(smem + 98304);         // 32*200*2 = 12800 B
    float*    sH  = (float*)   (smem + 111104);        // 32*196*4 = 25088 B
    _Float16* sHh = (_Float16*)(smem + 136192);        // 32*200*2 = 12800 B
    float*    sC  = (float*)   (smem + 148992);        // 640*4    =  2560 B  -> 151552 total

    const int tid  = threadIdx.x;
    const int wv   = tid >> 5;
    const int lane = tid & 31;

    // ---- one-time: weights + constants into LDS ----
    {
        const float4* g1 = (const float4*)w1p;
        float4*       s1 = (float4*)sW1;
        #pragma unroll
        for (int i = tid; i < 4608; i += 256) s1[i] = g1[i];
        const float4* g2 = (const float4*)w2p;
        float4*       s2 = (float4*)sW2;
        #pragma unroll
        for (int i = tid; i < 1536; i += 256) s2[i] = g2[i];
        for (int i = tid; i < 640; i += 256) {
            float v;
            if (i < 192)      v = b1[i];
            else if (i < 384) v = gamma[i - 192];
            else if (i < 576) v = beta[i - 384];
            else              v = b2[i - 576];
            sC[i] = v;
        }
    }
    __syncthreads();

    const int mt  = wv & 1;        // M-tile (0/1) for both GEMMs
    const int ntg = wv >> 1;       // 0..3
    const int m   = lane & 15;
    const int kh  = lane >> 4;

    for (int tile = blockIdx.x; tile < nTiles; tile += gridDim.x) {
        const int edgeBase = tile * ETILE;

        // ---- gather + f32->f16: x = [node[row] | node[col] | edge] into sX ----
        {
            int r   = tid >> 3;                 // edge row in tile: 0..31
            int sub = tid & 7;                  // 8 threads per edge
            int e   = edgeBase + r;
            int ec  = e < nEdges ? e : nEdges - 1;
            int row = eidx[ec];
            int col = eidx[nEdges + ec];
            const float4* srcA = (const float4*)(nodef + (size_t)row * NODE_DIM);
            const float4* srcB = (const float4*)(nodef + (size_t)col * NODE_DIM);
            const float4* srcE = (const float4*)(edgef + (size_t)ec  * EDGE_DIM);
            #pragma unroll
            for (int q = 0; q < 6; ++q) {
                int c = sub * 6 + q;            // 0..47 float4 chunks per row
                const float4* p = (c < 16) ? (srcA + c)
                                 : (c < 32) ? (srcB + (c - 16))
                                            : (srcE + (c - 32));
                float4 v = *p;
                v4h h;
                h[0] = (_Float16)v.x; h[1] = (_Float16)v.y;
                h[2] = (_Float16)v.z; h[3] = (_Float16)v.w;
                *(v4h*)(sX + r * XPITCH + c * 4) = h;
            }
        }
        __syncthreads();

        // ---- GEMM1: h = x @ W1 + b1  (wave owns mt x {ntg, ntg+4, ntg+8}) ----
        {
            const _Float16* arow = sX + (mt * 16 + m) * XPITCH;
            v8f acc[3] = {};
            #pragma unroll
            for (int ks = 0; ks < 6; ++ks) {
                const _Float16* ap = arow + ks * 32 + kh * 8;
                v8h alo = *(const v8h*)ap;
                v8h ahi = *(const v8h*)(ap + 16);
                v16h a = __builtin_shufflevector(alo, ahi, 0,1,2,3,4,5,6,7,8,9,10,11,12,13,14,15);
                #pragma unroll
                for (int t = 0; t < 3; ++t) {
                    int nt = ntg + t * 4;
                    const _Float16* bp = sW1 + ((ks * 12 + nt) * 32 + lane) * 16;
                    v8h blo = *(const v8h*)bp;
                    v8h bhi = *(const v8h*)(bp + 8);
                    v16h b = __builtin_shufflevector(blo, bhi, 0,1,2,3,4,5,6,7,8,9,10,11,12,13,14,15);
                    acc[t] = __builtin_amdgcn_wmma_f32_16x16x32_f16(
                        false, a, false, b, (short)0, acc[t], false, false);
                }
            }
            #pragma unroll
            for (int t = 0; t < 3; ++t) {
                int nt = ntg + t * 4;
                int n  = nt * 16 + (lane & 15);
                float bias = sC[n];
                #pragma unroll
                for (int j = 0; j < 8; ++j) {
                    int mrow = mt * 16 + j + ((lane >> 4) << 3);
                    sH[mrow * HPITCH + n] = acc[t][j] + bias;
                }
            }
        }
        __syncthreads();

        // ---- LayerNorm + ReLU -> f16 (8 lanes per row, shfl_xor reduction) ----
        {
            int r = tid >> 3, g = tid & 7;
            const float* hr = sH + r * HPITCH + g * 24;
            float s = 0.f, s2 = 0.f;
            #pragma unroll
            for (int q = 0; q < 24; ++q) { float v = hr[q]; s += v; s2 += v * v; }
            s  += __shfl_xor(s, 1, 8);  s  += __shfl_xor(s, 2, 8);  s  += __shfl_xor(s, 4, 8);
            s2 += __shfl_xor(s2, 1, 8); s2 += __shfl_xor(s2, 2, 8); s2 += __shfl_xor(s2, 4, 8);
            float mean = s * (1.0f / 192.0f);
            float var  = s2 * (1.0f / 192.0f) - mean * mean;
            float rstd = rsqrtf(var + 1e-5f);
            _Float16* hd = sHh + r * XPITCH + g * 24;
            #pragma unroll
            for (int q = 0; q < 24; ++q) {
                int cidx = g * 24 + q;
                float v = (hr[q] - mean) * rstd * sC[192 + cidx] + sC[384 + cidx];
                hd[q] = (_Float16)(v > 0.f ? v : 0.f);
            }
        }
        __syncthreads();

        // ---- GEMM2: out = relu_ln(h) @ W2 + b2 + edge_feats ----
        {
            int nt2 = wv >> 1;                       // 0..3
            const _Float16* hrow = sHh + (mt * 16 + m) * XPITCH;
            v8f acc = {};
            #pragma unroll
            for (int ks = 0; ks < 6; ++ks) {
                const _Float16* ap = hrow + ks * 32 + kh * 8;
                v8h alo = *(const v8h*)ap;
                v8h ahi = *(const v8h*)(ap + 16);
                v16h a = __builtin_shufflevector(alo, ahi, 0,1,2,3,4,5,6,7,8,9,10,11,12,13,14,15);
                const _Float16* bp = sW2 + ((ks * 4 + nt2) * 32 + lane) * 16;
                v8h blo = *(const v8h*)bp;
                v8h bhi = *(const v8h*)(bp + 8);
                v16h b = __builtin_shufflevector(blo, bhi, 0,1,2,3,4,5,6,7,8,9,10,11,12,13,14,15);
                acc = __builtin_amdgcn_wmma_f32_16x16x32_f16(
                    false, a, false, b, (short)0, acc, false, false);
            }
            int n = nt2 * 16 + (lane & 15);
            float bias = sC[576 + n];
            #pragma unroll
            for (int j = 0; j < 8; ++j) {
                int mrow = mt * 16 + j + ((lane >> 4) << 3);
                int e = edgeBase + mrow;
                if (e < nEdges) {
                    size_t o = (size_t)e * EDGE_DIM + n;
                    out[o] = acc[j] + bias + edgef[o];
                }
            }
        }
        __syncthreads();   // protect sX/sH/sHh reuse next iteration
    }
}

extern "C" void kernel_launch(void* const* d_in, const int* in_sizes, int n_in,
                              void* d_out, int out_size, void* d_ws, size_t ws_size,
                              hipStream_t stream) {
    const float* nodef = (const float*)d_in[0];
    const float* edgef = (const float*)d_in[1];
    const int*   eidx  = (const int*)d_in[2];
    const float* W1    = (const float*)d_in[3];
    const float* b1    = (const float*)d_in[4];
    const float* gamma = (const float*)d_in[5];
    const float* beta  = (const float*)d_in[6];
    const float* W2    = (const float*)d_in[7];
    const float* b2    = (const float*)d_in[8];
    float* out = (float*)d_out;

    const int E = in_sizes[1] / EDGE_DIM;   // edge_feats is [E, 64]

    _Float16* w1p = (_Float16*)d_ws;
    _Float16* w2p = (_Float16*)((char*)d_ws + 73728);

    pack_weights<<<144, 256, 0, stream>>>(W1, W2, w1p, w2p);

    const int nTiles = (E + ETILE - 1) / ETILE;
    int blocks = nTiles < 2048 ? nTiles : 2048;
    const size_t smemBytes = 151552;  // W1+W2(f16) + x(f16) + h(f32) + h(f16) + consts
    edge_update_kernel<<<blocks, 256, smemBytes, stream>>>(
        nodef, edgef, eidx, b1, gamma, beta, b2, w1p, w2p, out, E, nTiles);
}
